// ExpertPool_45346264711699
// MI455X (gfx1250) — matrix operand order repaired and checked by reading, not compiled, not run on hardware
//
#include <hip/hip_runtime.h>
#include <hip/hip_bf16.h>

typedef __attribute__((ext_vector_type(16))) __bf16 v16bf;
typedef __attribute__((ext_vector_type(8)))  float  v8f;

#define NE 8
#define NB 8192
#define ND 512
#define NH 256
#define NC 1000
#define LN_EPS 1e-5f
#define INV_TEMP 10.0f

union FragB { uint4 q[2]; v16bf v; };

__device__ __forceinline__ unsigned short f2bf(float f) {
  union { float f; unsigned u; } v; v.f = f;
  unsigned r = v.u + 0x7FFFu + ((v.u >> 16) & 1u);
  return (unsigned short)(r >> 16);
}

// ---------------- counting-sort helpers ----------------

__global__ void k_zero_meta(int* meta) {
  if (threadIdx.x < 24) meta[threadIdx.x] = 0;
}

__global__ void k_count(const int* __restrict__ ids, int* __restrict__ meta, int n) {
  int i = blockIdx.x * blockDim.x + threadIdx.x;
  if (i < n) atomicAdd(&meta[ids[i]], 1);
}

__global__ void k_scan(int* meta) {
  if (threadIdx.x == 0) {
    int acc = 0;
    for (int e = 0; e < NE; e++) { meta[8 + e] = acc; meta[16 + e] = acc; acc += meta[e]; }
  }
}

__global__ void k_place(const int* __restrict__ ids, int* __restrict__ meta,
                        int* __restrict__ perm, int n) {
  int i = blockIdx.x * blockDim.x + threadIdx.x;
  if (i < n) { int p = atomicAdd(&meta[16 + ids[i]], 1); perm[p] = i; }
}

// ---------------- dtype prep ----------------

__global__ void k_f32_to_bf16(const float* __restrict__ in,
                              unsigned short* __restrict__ out, int n) {
  for (int i = blockIdx.x * blockDim.x + threadIdx.x; i < n; i += gridDim.x * blockDim.x)
    out[i] = f2bf(in[i]);
}

// in: f32 [G][R][C]  ->  out: bf16 [G][C][R]
__global__ void k_transpose_bf16(const float* __restrict__ in,
                                 unsigned short* __restrict__ out,
                                 int G, int R, int C) {
  int n = G * R * C;
  for (int i = blockIdx.x * blockDim.x + threadIdx.x; i < n; i += gridDim.x * blockDim.x) {
    int g = i / (R * C); int rem = i - g * R * C;
    int r = rem / C;     int c = rem - r * C;
    out[((size_t)g * C + c) * R + r] = f2bf(in[i]);
  }
}

// l2-normalize each anchor row, write bf16 [C][D]
__global__ void k_anchor_norm(const float* __restrict__ anch,
                              unsigned short* __restrict__ out) {
  int wave = threadIdx.x >> 5, lane = threadIdx.x & 31;
  int c = blockIdx.x * 8 + wave;
  if (c >= NC) return;
  const float* row = anch + (size_t)c * ND;
  float vals[16]; float s = 0.f;
  for (int i = 0; i < 16; i++) { float v = row[lane + i * 32]; vals[i] = v; s += v * v; }
  for (int m = 1; m < 32; m <<= 1) s += __shfl_xor(s, m, 32);
  float inv = 1.0f / fmaxf(sqrtf(s), 1e-12f);
  for (int i = 0; i < 16; i++) out[(size_t)c * ND + lane + i * 32] = f2bf(vals[i] * inv);
}

__global__ void k_fill_neginf(float4* __restrict__ p, int n4) {
  float ninf = -__builtin_huge_valf();
  float4 v = make_float4(ninf, ninf, ninf, ninf);
  for (int i = blockIdx.x * blockDim.x + threadIdx.x; i < n4; i += gridDim.x * blockDim.x)
    p[i] = v;
}

// ---------------- weight staging: direct global -> LDS copy of one K-slice -----
// Short register live range (no cross-barrier values) => no scratch spills.
__device__ __forceinline__ void stageW(unsigned short* sWb,
                                       const unsigned short* __restrict__ wbase,
                                       int stride, int k0, int tid) {
  int c = tid & 3, hb = tid >> 2;
  #pragma unroll
  for (int j = 0; j < 4; j++) {
    uint4 t = *(const uint4*)(wbase + (size_t)(hb + j * 64) * stride + k0 + c * 8);
    *(uint4*)(sWb + (hb + j * 64) * 32 + c * 8) = t;
  }
}

// ---------------- expert MLP ----------------------------------------------------
// block = 256 threads = 8 waves = 4 wave-pairs. Each pair owns one 16-row M-tile
// (block M = 64); wave nh = wave&1 owns 8 of the 16 N-tiles (64 acc VGPRs).
// Weight K-slices double-buffered in LDS, one barrier per k-step.
__global__ void __launch_bounds__(256) k_moe_mlp(
    const unsigned short* __restrict__ xb,   // [B][D] bf16
    const int* __restrict__ perm, const int* __restrict__ meta,
    const unsigned short* __restrict__ w1t,  // [E][H][D] bf16
    const float* __restrict__ b1, const float* __restrict__ g1, const float* __restrict__ be1,
    const unsigned short* __restrict__ w2t,  // [E][H][H] bf16
    const float* __restrict__ b2, const float* __restrict__ g2, const float* __restrict__ be2,
    const unsigned short* __restrict__ w3t,  // [E][D][H] bf16
    const float* __restrict__ b3,
    float* __restrict__ ybuf)                // [B][D] f32, permuted order
{
  extern __shared__ char dsmem[];
  unsigned short* sH  = (unsigned short*)dsmem;            // 64*256 bf16 = 32KB
  unsigned short* sW  = (unsigned short*)(dsmem + 32768);  // 2 x (256x32) bf16 = 32KB
  float*          sLN = (float*)(dsmem + 65536);           // 64 rows x 2 halves x {s,q}

  int e   = blockIdx.y;
  int cnt = meta[e];
  int base = blockIdx.x * 64;
  if (cnt == 0 || base >= cnt) return;
  int off = meta[8 + e];

  int tid  = threadIdx.x;
  int wave = tid >> 5, lane = tid & 31;
  int lr = lane & 15, kh = lane >> 4;
  int p  = wave >> 1;        // M-tile (pair) index: 0..3
  int nh = wave & 1;         // N-half within pair

  int rowA  = base + p * 16 + lr;
  int rowAc = min(rowA, cnt - 1);
  int tok   = perm[off + rowAc];
  const unsigned short* xrow = xb + (size_t)tok * ND;

  // ---- GEMM1: [16 x 512] @ [512 x 256], this wave: N cols [nh*128, nh*128+128) ----
  v8f acc[8];
  for (int j = 0; j < 8; j++) for (int i = 0; i < 8; i++) acc[j][i] = 0.f;
  const unsigned short* wb1 = w1t + (size_t)e * NH * ND;
  for (int ks = 0; ks < 16; ks++) {
    int k0 = ks * 32;
    unsigned short* sWb = sW + (ks & 1) * (256 * 32);
    stageW(sWb, wb1, ND, k0, tid);
    __syncthreads();
    FragB a;
    a.q[0] = *(const uint4*)(xrow + k0 + kh * 16);
    a.q[1] = *(const uint4*)(xrow + k0 + kh * 16 + 8);
    const unsigned short* brb = sWb + (nh * 8 * 16 + lr) * 32 + kh * 16;
    FragB b0, b1f;
    b0.q[0] = *(const uint4*)brb; b0.q[1] = *(const uint4*)(brb + 8);
    #pragma unroll
    for (int j = 0; j < 8; j++) {
      FragB& cur = (j & 1) ? b1f : b0;
      FragB& nxt = (j & 1) ? b0 : b1f;
      if (j + 1 < 8) {
        const unsigned short* br = brb + (j + 1) * 512;
        nxt.q[0] = *(const uint4*)br; nxt.q[1] = *(const uint4*)(br + 8);
      }
      acc[j] = __builtin_amdgcn_wmma_f32_16x16x32_bf16(
          false, a.v, false, cur.v, (short)0, acc[j], false, false);
    }
  }

  // ---- bias + one-pass LayerNorm + ReLU -> sH ----
  {
    float bv[8], gv[8], ev[8];
    for (int j = 0; j < 8; j++) {
      int h = nh * 128 + j * 16 + lr;
      bv[j] = b1[e * NH + h]; gv[j] = g1[e * NH + h]; ev[j] = be1[e * NH + h];
    }
    for (int r = 0; r < 8; r++) {
      float s = 0.f, q = 0.f;
      for (int j = 0; j < 8; j++) {
        acc[j][r] += bv[j];
        s += acc[j][r]; q += acc[j][r] * acc[j][r];
      }
      for (int m = 1; m < 16; m <<= 1) { s += __shfl_xor(s, m, 16); q += __shfl_xor(q, m, 16); }
      if (lr == 0) {
        int m = p * 16 + r + 8 * kh;
        sLN[m * 4 + nh * 2 + 0] = s;
        sLN[m * 4 + nh * 2 + 1] = q;
      }
    }
    __syncthreads();
    for (int r = 0; r < 8; r++) {
      int m = p * 16 + r + 8 * kh;
      float S = sLN[m * 4 + 0] + sLN[m * 4 + 2];
      float Q = sLN[m * 4 + 1] + sLN[m * 4 + 3];
      float mean = S * (1.0f / NH);
      float var  = fmaxf(Q * (1.0f / NH) - mean * mean, 0.f);
      float rstd = rsqrtf(var + LN_EPS);
      for (int j = 0; j < 8; j++) {
        float v = (acc[j][r] - mean) * rstd * gv[j] + ev[j];
        sH[m * NH + nh * 128 + j * 16 + lr] = f2bf(fmaxf(v, 0.f));
      }
    }
  }

  const unsigned short* srowA = sH + (size_t)(p * 16 + lr) * NH;

  // ---- GEMM2: [16 x 256] @ [256 x 256] ----
  v8f acc2[8];
  for (int j = 0; j < 8; j++) for (int i = 0; i < 8; i++) acc2[j][i] = 0.f;
  const unsigned short* wb2 = w2t + (size_t)e * NH * NH;
  for (int ks = 0; ks < 8; ks++) {
    int k0 = ks * 32;
    unsigned short* sWb = sW + (ks & 1) * (256 * 32);
    stageW(sWb, wb2, NH, k0, tid);
    __syncthreads();
    FragB a;
    a.q[0] = *(const uint4*)(srowA + k0 + kh * 16);
    a.q[1] = *(const uint4*)(srowA + k0 + kh * 16 + 8);
    const unsigned short* brb = sWb + (nh * 8 * 16 + lr) * 32 + kh * 16;
    FragB b0, b1f;
    b0.q[0] = *(const uint4*)brb; b0.q[1] = *(const uint4*)(brb + 8);
    #pragma unroll
    for (int j = 0; j < 8; j++) {
      FragB& cur = (j & 1) ? b1f : b0;
      FragB& nxt = (j & 1) ? b0 : b1f;
      if (j + 1 < 8) {
        const unsigned short* br = brb + (j + 1) * 512;
        nxt.q[0] = *(const uint4*)br; nxt.q[1] = *(const uint4*)(br + 8);
      }
      acc2[j] = __builtin_amdgcn_wmma_f32_16x16x32_bf16(
          false, a.v, false, cur.v, (short)0, acc2[j], false, false);
    }
  }

  // ---- bias + one-pass LayerNorm + ReLU -> sH (overwrite own cols) ----
  {
    float bv[8], gv[8], ev[8];
    for (int j = 0; j < 8; j++) {
      int h = nh * 128 + j * 16 + lr;
      bv[j] = b2[e * NH + h]; gv[j] = g2[e * NH + h]; ev[j] = be2[e * NH + h];
    }
    __syncthreads();   // all GEMM2 A-reads of sH done before overwrite
    for (int r = 0; r < 8; r++) {
      float s = 0.f, q = 0.f;
      for (int j = 0; j < 8; j++) {
        acc2[j][r] += bv[j];
        s += acc2[j][r]; q += acc2[j][r] * acc2[j][r];
      }
      for (int m = 1; m < 16; m <<= 1) { s += __shfl_xor(s, m, 16); q += __shfl_xor(q, m, 16); }
      if (lr == 0) {
        int m = p * 16 + r + 8 * kh;
        sLN[m * 4 + nh * 2 + 0] = s;
        sLN[m * 4 + nh * 2 + 1] = q;
      }
    }
    __syncthreads();
    for (int r = 0; r < 8; r++) {
      int m = p * 16 + r + 8 * kh;
      float S = sLN[m * 4 + 0] + sLN[m * 4 + 2];
      float Q = sLN[m * 4 + 1] + sLN[m * 4 + 3];
      float mean = S * (1.0f / NH);
      float var  = fmaxf(Q * (1.0f / NH) - mean * mean, 0.f);
      float rstd = rsqrtf(var + LN_EPS);
      for (int j = 0; j < 8; j++) {
        float v = (acc2[j][r] - mean) * rstd * gv[j] + ev[j];
        sH[m * NH + nh * 128 + j * 16 + lr] = f2bf(fmaxf(v, 0.f));
      }
    }
  }

  // ---- GEMM3: [16 x 256] @ [256 x 512] in two N halves ----
  const unsigned short* wb3 = w3t + (size_t)e * ND * NH;
  for (int half = 0; half < 2; half++) {
    int nb = half * 256;
    v8f acc3[8];
    for (int j = 0; j < 8; j++) for (int i = 0; i < 8; i++) acc3[j][i] = 0.f;
    const unsigned short* wbh = wb3 + (size_t)nb * NH;
    for (int ks = 0; ks < 8; ks++) {
      int k0 = ks * 32;
      unsigned short* sWb = sW + (ks & 1) * (256 * 32);
      stageW(sWb, wbh, NH, k0, tid);
      __syncthreads();
      FragB a;
      a.q[0] = *(const uint4*)(srowA + k0 + kh * 16);
      a.q[1] = *(const uint4*)(srowA + k0 + kh * 16 + 8);
      const unsigned short* brb = sWb + (nh * 8 * 16 + lr) * 32 + kh * 16;
      FragB b0, b1f;
      b0.q[0] = *(const uint4*)brb; b0.q[1] = *(const uint4*)(brb + 8);
      #pragma unroll
      for (int j = 0; j < 8; j++) {
        FragB& cur = (j & 1) ? b1f : b0;
        FragB& nxt = (j & 1) ? b0 : b1f;
        if (j + 1 < 8) {
          const unsigned short* br = brb + (j + 1) * 512;
          nxt.q[0] = *(const uint4*)br; nxt.q[1] = *(const uint4*)(br + 8);
        }
        acc3[j] = __builtin_amdgcn_wmma_f32_16x16x32_bf16(
            false, a.v, false, cur.v, (short)0, acc3[j], false, false);
      }
    }
    float b3v[8];
    for (int j = 0; j < 8; j++) b3v[j] = b3[e * ND + nb + nh * 128 + j * 16 + lr];
    for (int r = 0; r < 8; r++) {
      int m = base + p * 16 + r + 8 * kh;
      if (m < cnt) {
        float* yr = ybuf + (size_t)(off + m) * ND;
        for (int j = 0; j < 8; j++) yr[nb + nh * 128 + j * 16 + lr] = acc3[j][r] + b3v[j];
      }
    }
  }
}

// ---------------- l2norm rows -> feats (f32 out + bf16 scratch) ----------------
__global__ void k_l2norm_feats(const float* __restrict__ ybuf, const int* __restrict__ perm,
                               float* __restrict__ feats, unsigned short* __restrict__ featsb) {
  int wave = threadIdx.x >> 5, lane = threadIdx.x & 31;
  int pos = blockIdx.x * 8 + wave;
  if (pos >= NB) return;
  const float* yr = ybuf + (size_t)pos * ND;
  float v[16]; float s = 0.f;
  for (int i = 0; i < 16; i++) { float t = yr[lane + i * 32]; v[i] = t; s += t * t; }
  for (int m = 1; m < 32; m <<= 1) s += __shfl_xor(s, m, 32);
  float inv = 1.0f / fmaxf(sqrtf(s), 1e-12f);
  int tok = perm[pos];
  float* fr = feats + (size_t)tok * ND;
  unsigned short* fb = featsb + (size_t)tok * ND;
  for (int i = 0; i < 16; i++) {
    float t = v[i] * inv;
    fr[lane + i * 32] = t;
    fb[lane + i * 32] = f2bf(t);
  }
}

// ---------------- per-expert class logits (classes sorted; no masking) ----------
// Wave pairs split K: wave kk = wave&1 holds 8 A-fragments (its K half) and
// computes a partial product; kk=1 dumps partials to LDS, kk=0 combines+stores.
__global__ void __launch_bounds__(256) k_logits_sorted(
    const unsigned short* __restrict__ featsb, const unsigned short* __restrict__ anchb,
    const int* __restrict__ perm, const int* __restrict__ tmeta,
    const int* __restrict__ cperm, const int* __restrict__ cmeta,
    float* __restrict__ logits)
{
  __shared__ unsigned short sC[2 * 16 * ND];   // 2 x 16KB class tile
  __shared__ int   sCls[2][16];
  __shared__ float sAcc[4 * 8 * 32];           // pair x r x lane partials

  int e    = blockIdx.y;
  int tcnt = tmeta[e];
  int base = blockIdx.x * 64;
  if (tcnt == 0 || base >= tcnt) return;
  int toff = tmeta[8 + e];
  int ccnt = cmeta[e];
  if (ccnt == 0) return;                       // all logits stay -inf
  int coff = cmeta[8 + e];

  int tid  = threadIdx.x;
  int wave = tid >> 5, lane = tid & 31;
  int lr = lane & 15, kh = lane >> 4;
  int p  = wave >> 1;        // M-tile (pair) index: 0..3
  int kk = wave & 1;         // K half

  // A fragments: 8 k-steps (this wave's K half) of the pair's 16 token rows
  int rowA  = base + p * 16 + lr;
  int rowAc = min(rowA, tcnt - 1);
  const unsigned short* arow = featsb + (size_t)perm[toff + rowAc] * ND;
  FragB a[8];
  #pragma unroll
  for (int j = 0; j < 8; j++) {
    int ks = kk * 8 + j;
    a[j].q[0] = *(const uint4*)(arow + ks * 32 + kh * 16);
    a[j].q[1] = *(const uint4*)(arow + ks * 32 + kh * 16 + 8);
  }
  int  tokr[8]; bool rowok[8];
  for (int r = 0; r < 8; r++) {
    int m = base + p * 16 + r + 8 * kh;
    rowok[r] = m < tcnt;
    tokr[r]  = perm[toff + min(m, tcnt - 1)];
  }

  int myrow = tid >> 6, myc16 = tid & 63;      // staging coords (4 rows per j-step)
  int nct = (ccnt + 15) >> 4;

  for (int ct = 0; ct < nct; ct++) {
    unsigned short* sCb = sC + (ct & 1) * (16 * ND);
    #pragma unroll
    for (int j = 0; j < 4; j++) {
      int cls = cperm[coff + min(ct * 16 + myrow + j * 4, ccnt - 1)];
      uint4 t = *(const uint4*)(anchb + (size_t)cls * ND + myc16 * 8);
      *(uint4*)(sCb + (myrow + j * 4) * ND + myc16 * 8) = t;
    }
    if (tid < 16) sCls[ct & 1][tid] = cperm[coff + min(ct * 16 + tid, ccnt - 1)];
    __syncthreads();
    v8f acc;
    for (int i = 0; i < 8; i++) acc[i] = 0.f;
    const unsigned short* cb16 = sCb + lr * ND + kk * 256 + kh * 16;
    FragB b0, b1f;
    b0.q[0] = *(const uint4*)cb16; b0.q[1] = *(const uint4*)(cb16 + 8);
    #pragma unroll
    for (int j = 0; j < 8; j++) {
      FragB& cur = (j & 1) ? b1f : b0;
      FragB& nxt = (j & 1) ? b0 : b1f;
      if (j + 1 < 8) {
        const unsigned short* br = cb16 + (j + 1) * 32;
        nxt.q[0] = *(const uint4*)br; nxt.q[1] = *(const uint4*)(br + 8);
      }
      acc = __builtin_amdgcn_wmma_f32_16x16x32_bf16(
          false, a[j].v, false, cur.v, (short)0, acc, false, false);
    }
    if (kk == 1) {
      for (int r = 0; r < 8; r++) sAcc[(p * 8 + r) * 32 + lane] = acc[r];
    }
    __syncthreads();
    if (kk == 0) {
      bool cok = (ct * 16 + lr) < ccnt;
      int  cls = sCls[ct & 1][lr];
      if (cok) {
        for (int r = 0; r < 8; r++) {
          float tot = acc[r] + sAcc[(p * 8 + r) * 32 + lane];
          if (rowok[r]) logits[(size_t)tokr[r] * NC + cls] = tot * INV_TEMP;
        }
      }
    }
  }
}

// ---------------- launch ----------------
extern "C" void kernel_launch(void* const* d_in, const int* in_sizes, int n_in,
                              void* d_out, int out_size, void* d_ws, size_t ws_size,
                              hipStream_t stream) {
  const float* x    = (const float*)d_in[0];
  const int*   eid  = (const int*)d_in[1];
  const float* anch = (const float*)d_in[2];
  const int*   cexp = (const int*)d_in[3];
  const float* W1   = (const float*)d_in[4];
  const float* b1   = (const float*)d_in[5];
  const float* g1   = (const float*)d_in[6];
  const float* be1  = (const float*)d_in[7];
  const float* W2   = (const float*)d_in[8];
  const float* b2   = (const float*)d_in[9];
  const float* g2   = (const float*)d_in[10];
  const float* be2  = (const float*)d_in[11];
  const float* W3   = (const float*)d_in[12];
  const float* b3   = (const float*)d_in[13];

  float* out    = (float*)d_out;
  float* logits = out;                       // [B*C]
  float* feats  = out + (size_t)NB * NC;     // [B*D]

  char* ws = (char*)d_ws;
  size_t o = 0;
  auto alloc = [&](size_t bytes) { char* p = ws + o; o += (bytes + 255) & ~(size_t)255; return p; };
  unsigned short* xb     = (unsigned short*)alloc((size_t)NB * ND * 2);
  unsigned short* w1t    = (unsigned short*)alloc((size_t)NE * NH * ND * 2);
  unsigned short* w2t    = (unsigned short*)alloc((size_t)NE * NH * NH * 2);
  unsigned short* w3t    = (unsigned short*)alloc((size_t)NE * ND * NH * 2);
  unsigned short* anchb  = (unsigned short*)alloc((size_t)NC * ND * 2);
  unsigned short* featsb = (unsigned short*)alloc((size_t)NB * ND * 2);
  float*          ybuf   = (float*)alloc((size_t)NB * ND * 4);
  int*            perm   = (int*)alloc((size_t)NB * 4);
  int*            tmeta  = (int*)alloc(64 * 4);
  int*            cperm  = (int*)alloc((size_t)NC * 4);
  int*            cmeta  = (int*)alloc(64 * 4);

  // token counting-sort by expert
  hipLaunchKernelGGL(k_zero_meta, dim3(1), dim3(32), 0, stream, tmeta);
  hipLaunchKernelGGL(k_count, dim3(NB / 256), dim3(256), 0, stream, eid, tmeta, NB);
  hipLaunchKernelGGL(k_scan, dim3(1), dim3(32), 0, stream, tmeta);
  hipLaunchKernelGGL(k_place, dim3(NB / 256), dim3(256), 0, stream, eid, tmeta, perm, NB);

  // class counting-sort by expert
  hipLaunchKernelGGL(k_zero_meta, dim3(1), dim3(32), 0, stream, cmeta);
  hipLaunchKernelGGL(k_count, dim3(4), dim3(256), 0, stream, cexp, cmeta, NC);
  hipLaunchKernelGGL(k_scan, dim3(1), dim3(32), 0, stream, cmeta);
  hipLaunchKernelGGL(k_place, dim3(4), dim3(256), 0, stream, cexp, cmeta, cperm, NC);

  // dtype prep
  hipLaunchKernelGGL(k_f32_to_bf16, dim3(2048), dim3(256), 0, stream, x, xb, NB * ND);
  hipLaunchKernelGGL(k_transpose_bf16, dim3(1024), dim3(256), 0, stream, W1, w1t, NE, ND, NH);
  hipLaunchKernelGGL(k_transpose_bf16, dim3(512),  dim3(256), 0, stream, W2, w2t, NE, NH, NH);
  hipLaunchKernelGGL(k_transpose_bf16, dim3(1024), dim3(256), 0, stream, W3, w3t, NE, NH, ND);
  hipLaunchKernelGGL(k_anchor_norm, dim3((NC + 7) / 8), dim3(256), 0, stream, anch, anchb);

  // fused per-expert MLP (dynamic LDS: 32KB sH + 2x16KB weight slices + 1KB LN)
  hipLaunchKernelGGL(k_moe_mlp, dim3(128, NE), dim3(256), 66560, stream,
                     xb, perm, tmeta, w1t, b1, g1, be1, w2t, b2, g2, be2, w3t, b3, ybuf);

  hipLaunchKernelGGL(k_l2norm_feats, dim3(NB / 8), dim3(256), 0, stream, ybuf, perm, feats, featsb);

  // logits: -inf fill, then per-expert (token x class) similarity
  hipLaunchKernelGGL(k_fill_neginf, dim3(2048), dim3(256), 0, stream,
                     (float4*)logits, NB * NC / 4);
  hipLaunchKernelGGL(k_logits_sorted, dim3(NB / 64, NE), dim3(256), 0, stream,
                     featsb, anchb, perm, tmeta, cperm, cmeta, logits);
}
